// MLPEdgeClassifier_90649579749888
// MI455X (gfx1250) — compile-verified, matrix operand order checked by reading, not run on hardware
//
#include <hip/hip_runtime.h>
#include <math.h>

typedef __attribute__((ext_vector_type(8)))  _Float16 v8h;
typedef __attribute__((ext_vector_type(16))) _Float16 v16h;
typedef __attribute__((ext_vector_type(8)))  float    v8f;

#define N_NODES   50000
#define N_EDGES   300000
#define NODE_DIM  128
#define HIDDEN    256
#define HC        512          // combined hidden (two MLPs fused into one GEMM)
#define KDIM      256          // 2*NODE_DIM input features
#define EPB       64           // edges per block
#define ROW_H     (KDIM + 8)   // LDS A-tile row stride in halves (+16B pad vs banks)

// d_ws layout:
//   [0, WC_BYTES)          : f16 combined weights, WMMA-B fragment order
//   then scale[512], shift[512], w2c[512] (f32)
#define WC_HALF_ELEMS (32 * 8 * 32 * 16)   // ntile * kstep * lane * half
#define WC_BYTES      (WC_HALF_ELEMS * 2)

// ---------------------------------------------------------------------------
// Pack Wc = [W1 | roll(W1,-16,axis=0)] (256x512 f32 -> f16) into WMMA-B
// fragment order. Fragment (ntile, kstep): lane holds 16 halves; lane<16 is
// column N=ntile*16+lane with K offsets {0..7,16..23}, lane>=16 same N set
// with K offsets {8..15,24..31} (mirrors the ISA 16-bit A 16x32 layout).
// ---------------------------------------------------------------------------
__global__ void prep_weights(const float* __restrict__ W1,
                             _Float16* __restrict__ wc) {
    int tid = blockIdx.x * 256 + threadIdx.x;
    if (tid >= WC_HALF_ELEMS) return;
    int h     = tid & 15;
    int lane  = (tid >> 4) & 31;
    int kstep = (tid >> 9) & 7;
    int nt    = tid >> 12;
    int koff  = (h < 8 ? h : h + 8) + (lane < 16 ? 0 : 8);
    int kk    = kstep * 32 + koff;          // 0..255
    int n     = nt * 16 + (lane & 15);      // 0..511
    float v;
    if (n < HIDDEN) v = W1[kk * HIDDEN + n];                       // mlp1
    else            v = W1[((kk + 16) & 255) * HIDDEN + (n - HIDDEN)]; // mlp2 (roll fused)
    wc[tid] = (_Float16)v;
}

// Fold b1 + BN(eval) into per-column affine; duplicate W2 for both halves.
__global__ void prep_params(const float* __restrict__ b1, const float* __restrict__ g,
                            const float* __restrict__ be, const float* __restrict__ mu,
                            const float* __restrict__ var, const float* __restrict__ W2,
                            float* __restrict__ scale, float* __restrict__ shift,
                            float* __restrict__ w2c) {
    int n = blockIdx.x * 256 + threadIdx.x;
    if (n >= HC) return;
    int m = n & (HIDDEN - 1);
    float sc = g[m] * rsqrtf(var[m] + 1e-5f);
    scale[n] = sc;
    shift[n] = (b1[m] - mu[m]) * sc + be[m];
    w2c[n]   = W2[m];
}

// ---------------------------------------------------------------------------
// Main: per block, gather 64 edges' features (f32->f16) into LDS, then a
// [64 x 256] x [256 x 512] WMMA GEMM (each wave owns 4 N-tiles, all 4 M-tiles),
// BN+ReLU+W2 epilogue with half-wave shuffle reduction + LDS f32 atomics.
// ---------------------------------------------------------------------------
__global__ __launch_bounds__(256) void edge_mlp_kernel(
    const float* __restrict__ node_feat, const int* __restrict__ eidx,
    const _Float16* __restrict__ wc, const float* __restrict__ scale,
    const float* __restrict__ shift, const float* __restrict__ w2c,
    const float* __restrict__ b2, float* __restrict__ out)
{
    __shared__ _Float16 At[EPB][ROW_H];
    __shared__ float    esum[EPB];

    const int tid = threadIdx.x;
    if (tid < EPB) esum[tid] = 0.0f;

    // ---- gather + convert: 4 threads per edge, 64 f32 each ----
    const int e0   = blockIdx.x * EPB;
    const int eLoc = tid >> 2;
    const int q    = tid & 3;
    int e = e0 + eLoc; if (e > N_EDGES - 1) e = N_EDGES - 1;   // clamp tail
    const int node = (q < 2) ? eidx[e] : eidx[N_EDGES + e];    // row / col endpoint
    const float* src = node_feat + (size_t)node * NODE_DIM + (size_t)(q & 1) * 64;
    _Float16* dst = &At[eLoc][q * 64];
    #pragma unroll
    for (int i = 0; i < 8; ++i) {
        const float4 a = *(const float4*)(src + i * 8);
        const float4 b = *(const float4*)(src + i * 8 + 4);
        v8h hv = { (_Float16)a.x, (_Float16)a.y, (_Float16)a.z, (_Float16)a.w,
                   (_Float16)b.x, (_Float16)b.y, (_Float16)b.z, (_Float16)b.w };
        *(v8h*)(dst + i * 8) = hv;
    }
    __syncthreads();

    // ---- WMMA GEMM ----
    const int wave  = tid >> 5;
    const int lane  = tid & 31;
    const int lhalf = lane >> 4;       // 0|1 -> K-offset group for A/B frags
    const int l16   = lane & 15;

    v8f acc[4][4];
    #pragma unroll
    for (int mt = 0; mt < 4; ++mt)
        #pragma unroll
        for (int j = 0; j < 4; ++j) acc[mt][j] = (v8f)0.0f;

    #pragma unroll
    for (int ks = 0; ks < 8; ++ks) {
        v16h afrag[4];
        #pragma unroll
        for (int mt = 0; mt < 4; ++mt) {
            const _Float16* ap = &At[mt * 16 + l16][ks * 32 + lhalf * 8];
            v8h lo = *(const v8h*)ap;          // K offsets +0..7  (or +8..15)
            v8h hi = *(const v8h*)(ap + 16);   // K offsets +16..23 (or +24..31)
            afrag[mt] = __builtin_shufflevector(lo, hi,
                0, 1, 2, 3, 4, 5, 6, 7, 8, 9, 10, 11, 12, 13, 14, 15);
        }
        v16h bfrag[4];
        #pragma unroll
        for (int j = 0; j < 4; ++j) {
            const _Float16* bp = wc + (size_t)(((wave * 4 + j) * 8 + ks) * 32 + lane) * 16;
            v8h lo = *(const v8h*)bp;
            v8h hi = *(const v8h*)(bp + 8);
            bfrag[j] = __builtin_shufflevector(lo, hi,
                0, 1, 2, 3, 4, 5, 6, 7, 8, 9, 10, 11, 12, 13, 14, 15);
        }
        #pragma unroll
        for (int mt = 0; mt < 4; ++mt)
            #pragma unroll
            for (int j = 0; j < 4; ++j)
                acc[mt][j] = __builtin_amdgcn_wmma_f32_16x16x32_f16(
                    false, afrag[mt], false, bfrag[j],
                    (short)0, acc[mt][j], false, false);
    }

    // ---- epilogue: BN + ReLU + dot(W2), reduce over 512 columns ----
    float sc[4], sh[4], w2[4];
    #pragma unroll
    for (int j = 0; j < 4; ++j) {
        int n = (wave * 4 + j) * 16 + l16;
        sc[j] = scale[n]; sh[j] = shift[n]; w2[j] = w2c[n];
    }
    #pragma unroll
    for (int mt = 0; mt < 4; ++mt) {
        #pragma unroll
        for (int r = 0; r < 8; ++r) {        // C/D: VGPR r -> M = r (+8 hi half)
            float s = 0.0f;
            #pragma unroll
            for (int j = 0; j < 4; ++j) {
                float h = acc[mt][j][r] * sc[j] + sh[j];
                h = fmaxf(h, 0.0f);
                s += h * w2[j];
            }
            s += __shfl_xor(s, 1, 32);
            s += __shfl_xor(s, 2, 32);
            s += __shfl_xor(s, 4, 32);
            s += __shfl_xor(s, 8, 32);
            if (l16 == 0)
                atomicAdd(&esum[mt * 16 + r + lhalf * 8], s);   // ds_add_f32
        }
    }
    __syncthreads();

    if (tid < EPB) {
        int ee = e0 + tid;
        if (ee < N_EDGES) {
            float v = 0.5f * esum[tid] + b2[0];
            out[ee] = 1.0f / (1.0f + __expf(-v));
        }
    }
}

extern "C" void kernel_launch(void* const* d_in, const int* in_sizes, int n_in,
                              void* d_out, int out_size, void* d_ws, size_t ws_size,
                              hipStream_t stream) {
    const float* node_feat = (const float*)d_in[0];
    const int*   eidx      = (const int*)  d_in[1];
    const float* W1        = (const float*)d_in[2];
    const float* b1        = (const float*)d_in[3];
    const float* gamma     = (const float*)d_in[4];
    const float* beta      = (const float*)d_in[5];
    const float* mean      = (const float*)d_in[6];
    const float* var       = (const float*)d_in[7];
    const float* W2        = (const float*)d_in[8];
    const float* b2        = (const float*)d_in[9];

    _Float16* wc   = (_Float16*)d_ws;
    float*   scale = (float*)((char*)d_ws + WC_BYTES);
    float*   shift = scale + HC;
    float*   w2c   = shift + HC;

    prep_weights<<<WC_HALF_ELEMS / 256, 256, 0, stream>>>(W1, wc);
    prep_params<<<(HC + 255) / 256, 256, 0, stream>>>(b1, gamma, beta, mean, var, W2,
                                                      scale, shift, w2c);

    const int nblk = (N_EDGES + EPB - 1) / EPB;
    edge_mlp_kernel<<<nblk, 256, 0, stream>>>(node_feat, eidx, wc, scale, shift,
                                              w2c, b2, (float*)d_out);
}
